// MambaSeg2D_51857435132182
// MI455X (gfx1250) — matrix operand order repaired and checked by reading, not compiled, not run on hardware
//
#include <hip/hip_runtime.h>
#include <hip/hip_bf16.h>

// ---------------------------------------------------------------------------
// MambaSeg2D on MI455X (gfx1250): bf16 WMMA GEMMs (packed fragments, guard-
// free, 4 M-tiles per wave) + f32 VALU selective scan.
// L = 4096 tokens, D_MODEL=96, D_INNER=192, D_STATE=16, DT_RANK=6.
// ---------------------------------------------------------------------------

#define L_TOK   4096
#define DMODEL  96
#define DINNER  192
#define DSTATE  16
#define DTRANK  6
#define XDBL_NP 48          // x_dbl row stride, 38 padded to 48
#define DT_KP   32          // dt_proj K, 6 padded to 32

typedef __attribute__((ext_vector_type(16))) __bf16       v16bf;
typedef __attribute__((ext_vector_type(8)))  float        v8f;
typedef __attribute__((ext_vector_type(4)))  unsigned int u32x4;

union Frag16 { v16bf v; u32x4 q[2]; };

static __device__ __forceinline__ __bf16 f2bf(float f) {
    unsigned u = __builtin_bit_cast(unsigned, f);
    unsigned r = u + 0x7fffu + ((u >> 16) & 1u);          // round-to-nearest-even
    unsigned short h = (unsigned short)(r >> 16);
    return __builtin_bit_cast(__bf16, h);
}

static __device__ __forceinline__ float silu(float v) {
    return v * (1.0f / (1.0f + __expf(-v)));
}

// ---------------------------------------------------------------------------
// Pack f32 weight matrix (Kreal x Nreal, row stride ldb) into WMMA-native
// bf16 B fragments, zero-padded to Kp x Np (Kp%32==0, Np%16==0).
// Fragment layout (CDNA5 ISA 7.12.2, 16-bit B 32x16):
//   lane = 16*half + col ; VGPR p holds K = 2p + 8*half + 8*(p>>2) (packed pairs)
// dst[((kt*TN + nt)*32 + lane)*16 + j]
// ---------------------------------------------------------------------------
__global__ void pack_b_kernel(const float* __restrict__ src, int ldb,
                              int Kreal, int Nreal, int Kp, int Np,
                              __bf16* __restrict__ dst) {
    int idx = blockIdx.x * blockDim.x + threadIdx.x;
    int TN = Np >> 4;
    int total = (Kp >> 5) * TN * 32;
    if (idx >= total) return;
    int lane = idx & 31;
    int t    = idx >> 5;
    int nt   = t % TN;
    int kt   = t / TN;
    int half = lane >> 4;
    int col  = nt * 16 + (lane & 15);

    Frag16 f;
    #pragma unroll
    for (int p = 0; p < 8; ++p) {
        #pragma unroll
        for (int o = 0; o < 2; ++o) {
            int k = kt * 32 + 2 * p + 8 * half + 8 * (p >> 2) + o;
            float x = (k < Kreal && col < Nreal) ? src[k * ldb + col] : 0.0f;
            f.v[2 * p + o] = f2bf(x);
        }
    }
    u32x4* out = (u32x4*)(dst + (size_t)idx * 16);
    out[0] = f.q[0];
    out[1] = f.q[1];
}

// f32 -> bf16 bulk convert (for the network input x).
__global__ void cvt_bf16_kernel(const float* __restrict__ src,
                                __bf16* __restrict__ dst, int n) {
    int i = blockIdx.x * blockDim.x + threadIdx.x;
    if (i < n) dst[i] = f2bf(src[i]);
}

// ---------------------------------------------------------------------------
// Guard-free bf16 WMMA GEMM: C[M,N] = A[M,K] @ B[K,N], K = 32*Ktiles,
// N%16==0, M%64==0. One wave computes a 64x16 C strip: 4 M-tiles sharing one
// B fragment -> 4 v_wmma_f32_16x16x32_bf16 per K-step.
// A: bf16, row-major, lda elements. A fragment per lane = two contiguous
//    16-byte chunks at k = 8*half and k = 16 + 8*half  (ISA 16-bit A 16x32).
// Bp: pre-packed fragments (see pack_b_kernel).
// epilogue: 0 = none, 1 = softplus(x + bias[col]).  Optional bf16 C copy.
// ---------------------------------------------------------------------------
__global__ void gemm_bf16_wmma(const __bf16* __restrict__ A, int lda,
                               const __bf16* __restrict__ Bp,
                               float* __restrict__ C, int ldc,
                               __bf16* __restrict__ Cbf,
                               int M, int N, int Ktiles,
                               const float* __restrict__ bias, int epilogue) {
    const int wave = threadIdx.x >> 5;
    const int lane = threadIdx.x & 31;
    const int half = lane >> 4;
    const int sub  = lane & 15;

    const int TN   = N >> 4;
    const int tile = blockIdx.x * 4 + wave;
    if (tile >= (M >> 6) * TN) return;
    const int tm4 = tile / TN;          // 64-row strip index
    const int tn  = tile % TN;

    const int col_g = tn * 16 + sub;

    v8f acc[4] = {{}, {}, {}, {}};

    for (int kt = 0; kt < Ktiles; ++kt) {
        // B fragment: 32 contiguous bytes per lane (2x global_load_b128).
        const u32x4* bp =
            (const u32x4*)(Bp + ((size_t)(kt * TN + tn) * 32 + lane) * 16);
        Frag16 b;
        b.q[0] = bp[0];
        b.q[1] = bp[1];

        __builtin_prefetch(Bp + ((size_t)((kt + 1) * TN + tn) * 32 + lane) * 16, 0, 0);

        #pragma unroll
        for (int i = 0; i < 4; ++i) {
            const __bf16* ar =
                A + (size_t)(tm4 * 64 + i * 16 + sub) * lda + kt * 32 + 8 * half;
            Frag16 a;
            a.q[0] = *(const u32x4*)(ar);        // k = 8h .. 8h+7
            a.q[1] = *(const u32x4*)(ar + 16);   // k = 16+8h .. 16+8h+7
            acc[i] = __builtin_amdgcn_wmma_f32_16x16x32_bf16(
                         false, a.v, false, b.v, (short)0, acc[i], false, false);
        }
    }

    float bcol = (epilogue == 1) ? bias[col_g] : 0.0f;
    #pragma unroll
    for (int i = 0; i < 4; ++i) {
        #pragma unroll
        for (int r = 0; r < 8; ++r) {
            int mrow = tm4 * 64 + i * 16 + r + 8 * half;
            float v = acc[i][r];
            if (epilogue == 1) {                 // softplus(v + bias[col])
                float x = v + bcol;
                v = (x > 20.0f) ? x : __logf(1.0f + __expf(x));
            }
            C[(size_t)mrow * ldc + col_g] = v;
            if (Cbf) Cbf[(size_t)mrow * ldc + col_g] = f2bf(v);
        }
    }
}

// ---------------------------------------------------------------------------
// Depthwise 3x3 conv (SAME) + bias + SiLU on channels [0,192) of xz.
// Writes f32 + bf16 copies of u, and seeds y_sum = 4 * D_skip[c] * u.
// ---------------------------------------------------------------------------
__global__ void conv_silu_kernel(const float* __restrict__ xz,
                                 const float* __restrict__ conv_w,
                                 const float* __restrict__ conv_b,
                                 const float* __restrict__ D_skip,
                                 float* __restrict__ xconv,
                                 __bf16* __restrict__ xconv_bf,
                                 float* __restrict__ y_sum) {
    int idx = blockIdx.x * blockDim.x + threadIdx.x;
    if (idx >= L_TOK * DINNER) return;
    int c   = idx % DINNER;
    int pos = idx / DINNER;
    int hh  = pos >> 6, ww = pos & 63;

    float acc = conv_b[c];
    #pragma unroll
    for (int di = 0; di < 3; ++di) {
        int h2 = hh + di - 1;
        if ((unsigned)h2 >= 64u) continue;
        #pragma unroll
        for (int dj = 0; dj < 3; ++dj) {
            int w2 = ww + dj - 1;
            if ((unsigned)w2 >= 64u) continue;
            acc += xz[(h2 * 64 + w2) * (2 * DINNER) + c] * conv_w[c * 9 + di * 3 + dj];
        }
    }
    float u = silu(acc);
    xconv[idx]    = u;
    xconv_bf[idx] = f2bf(u);
    y_sum[idx]    = 4.0f * D_skip[c] * u;
}

// ---------------------------------------------------------------------------
// Four-direction selective scan. One 16-lane half-wave per (direction,channel);
// lane n = state index. h <- exp(dt*A)h + dt*B*u ; y += reduce_n(h*C).
// 768 groups = 48 blocks x 256 threads, each walking 4096 steps.
// ---------------------------------------------------------------------------
__global__ void scan_kernel(const float* __restrict__ xconv,
                            const float* __restrict__ xdbl,
                            const float* __restrict__ dt_full,
                            const float* __restrict__ A_log,
                            float* __restrict__ y_sum) {
    int group = blockIdx.x * (blockDim.x >> 4) + (threadIdx.x >> 4);
    int n   = threadIdx.x & 15;
    int dir = group / DINNER;
    int d   = group % DINNER;

    float A_dn = -__expf(A_log[d * DSTATE + n]);
    float h = 0.0f;

    for (int t = 0; t < L_TOK; ++t) {
        int l;
        if (dir == 0)      l = t;
        else if (dir == 1) l = (L_TOK - 1) - t;
        else {
            int tt = (dir == 2) ? t : (L_TOK - 1) - t;   // col-major traversal
            l = ((tt & 63) << 6) | (tt >> 6);            // (w,h) -> h*64+w
        }
        float dtv = dt_full[l * DINNER + d];
        float uv  = xconv  [l * DINNER + d];
        float Bv  = xdbl[l * XDBL_NP + DTRANK + n];
        float Cv  = xdbl[l * XDBL_NP + DTRANK + DSTATE + n];

        h = __expf(dtv * A_dn) * h + dtv * Bv * uv;
        float p = h * Cv;
        p += __shfl_xor(p, 8);
        p += __shfl_xor(p, 4);
        p += __shfl_xor(p, 2);
        p += __shfl_xor(p, 1);
        if (n == 0) atomicAdd(&y_sum[l * DINNER + d], p);
    }
}

// ---------------------------------------------------------------------------
// LayerNorm over 192 channels + gate by silu(z): g = LN(y)*silu(z) -> bf16.
// ---------------------------------------------------------------------------
__global__ void ln_gate_kernel(const float* __restrict__ y_sum,
                               const float* __restrict__ xz,
                               const float* __restrict__ ln_g,
                               const float* __restrict__ ln_b,
                               __bf16* __restrict__ g_bf) {
    __shared__ float sm[256];
    int l = blockIdx.x;
    int d = threadIdx.x;
    bool ok = d < DINNER;
    float v = ok ? y_sum[l * DINNER + d] : 0.0f;

    sm[d] = v;
    __syncthreads();
    for (int s = 128; s > 0; s >>= 1) {
        if (d < s) sm[d] += sm[d + s];
        __syncthreads();
    }
    float mean = sm[0] * (1.0f / DINNER);
    __syncthreads();

    float dv = ok ? (v - mean) : 0.0f;
    sm[d] = dv * dv;
    __syncthreads();
    for (int s = 128; s > 0; s >>= 1) {
        if (d < s) sm[d] += sm[d + s];
        __syncthreads();
    }
    float var = sm[0] * (1.0f / DINNER);

    if (ok) {
        float z = xz[l * (2 * DINNER) + DINNER + d];
        float y = dv * rsqrtf(var + 1e-5f) * ln_g[d] + ln_b[d];
        g_bf[l * DINNER + d] = f2bf(y * silu(z));
    }
}

// ---------------------------------------------------------------------------
static inline void launch_gemm(const __bf16* A, int lda, const __bf16* Bp,
                               float* C, int ldc, __bf16* Cbf,
                               int M, int N, int Ktiles,
                               const float* bias, int epi, hipStream_t s) {
    int tiles = (M >> 6) * (N >> 4);
    gemm_bf16_wmma<<<(tiles + 3) / 4, 128, 0, s>>>(A, lda, Bp, C, ldc, Cbf,
                                                   M, N, Ktiles, bias, epi);
}

static inline void launch_pack(const float* src, int ldb, int Kr, int Nr,
                               int Kp, int Np, __bf16* dst, hipStream_t s) {
    int total = (Kp >> 5) * (Np >> 4) * 32;
    pack_b_kernel<<<(total + 255) / 256, 256, 0, s>>>(src, ldb, Kr, Nr, Kp, Np, dst);
}

extern "C" void kernel_launch(void* const* d_in, const int* in_sizes, int n_in,
                              void* d_out, int out_size, void* d_ws, size_t ws_size,
                              hipStream_t stream) {
    const float* x         = (const float*)d_in[0];   // (4096, 96)
    const float* in_proj_w = (const float*)d_in[1];   // (96, 384)
    const float* conv_w    = (const float*)d_in[2];   // (192,1,3,3)
    const float* conv_b    = (const float*)d_in[3];   // (192,)
    const float* x_proj_w  = (const float*)d_in[4];   // (192, 38)
    const float* dt_proj_w = (const float*)d_in[5];   // (6, 192)
    const float* dt_proj_b = (const float*)d_in[6];   // (192,)
    const float* A_log     = (const float*)d_in[7];   // (192, 16)
    const float* D_skip    = (const float*)d_in[8];   // (192,)
    const float* out_proj_w= (const float*)d_in[9];   // (192, 96)
    const float* ln_g      = (const float*)d_in[10];  // (192,)
    const float* ln_b      = (const float*)d_in[11];  // (192,)
    float* out = (float*)d_out;                       // (4096, 96)

    // ---- workspace carve-up -------------------------------------------------
    float* ws = (float*)d_ws;
    float* xz      = ws;                                    // 4096*384
    float* xconv   = xz      + (size_t)L_TOK * 2 * DINNER;  // 4096*192
    float* y_sum   = xconv   + (size_t)L_TOK * DINNER;      // 4096*192
    float* xdbl    = y_sum   + (size_t)L_TOK * DINNER;      // 4096*48
    float* dt_full = xdbl    + (size_t)L_TOK * XDBL_NP;     // 4096*192
    __bf16* bfws   = (__bf16*)(dt_full + (size_t)L_TOK * DINNER);
    __bf16* x_bf      = bfws;                               // 4096*96
    __bf16* xconv_bf  = x_bf     + (size_t)L_TOK * DMODEL;  // 4096*192
    __bf16* xdbl_bf   = xconv_bf + (size_t)L_TOK * DINNER;  // 4096*48
    __bf16* g_bf      = xdbl_bf  + (size_t)L_TOK * XDBL_NP; // 4096*192
    __bf16* pw_in     = g_bf     + (size_t)L_TOK * DINNER;  // 96*384
    __bf16* pw_xp     = pw_in    + (size_t)DMODEL * 2 * DINNER;   // 192*48
    __bf16* pw_dt     = pw_xp    + (size_t)DINNER * XDBL_NP;      // 32*192
    __bf16* pw_out    = pw_dt    + (size_t)DT_KP * DINNER;        // 192*96

    // ---- 0. pack weights into WMMA-native bf16 fragments --------------------
    launch_pack(in_proj_w, 2 * DINNER, DMODEL, 2 * DINNER, DMODEL, 2 * DINNER,
                pw_in, stream);
    launch_pack(x_proj_w, DTRANK + 2 * DSTATE, DINNER, DTRANK + 2 * DSTATE,
                DINNER, XDBL_NP, pw_xp, stream);
    launch_pack(dt_proj_w, DINNER, DTRANK, DINNER, DT_KP, DINNER, pw_dt, stream);
    launch_pack(out_proj_w, DMODEL, DINNER, DMODEL, DINNER, DMODEL, pw_out, stream);

    // ---- 0b. input x -> bf16 ------------------------------------------------
    cvt_bf16_kernel<<<(L_TOK * DMODEL + 255) / 256, 256, 0, stream>>>(
        x, x_bf, L_TOK * DMODEL);

    // ---- 1. in_proj: xz = x @ in_proj_w  (WMMA, K=96, N=384) ----------------
    launch_gemm(x_bf, DMODEL, pw_in, xz, 2 * DINNER, nullptr,
                L_TOK, 2 * DINNER, DMODEL / 32, nullptr, 0, stream);

    // ---- 2. depthwise conv + SiLU; seeds y_sum = 4*D_skip*u -----------------
    conv_silu_kernel<<<(L_TOK * DINNER + 255) / 256, 256, 0, stream>>>(
        xz, conv_w, conv_b, D_skip, xconv, xconv_bf, y_sum);

    // ---- 3. x_proj: xdbl = u @ x_proj_w  (WMMA, K=192, N=48 padded) ---------
    launch_gemm(xconv_bf, DINNER, pw_xp, xdbl, XDBL_NP, xdbl_bf,
                L_TOK, XDBL_NP, DINNER / 32, nullptr, 0, stream);

    // ---- 4. dt = softplus(dt_raw @ dt_proj_w + b) (WMMA, K padded to 32) ----
    launch_gemm(xdbl_bf, XDBL_NP, pw_dt, dt_full, DINNER, nullptr,
                L_TOK, DINNER, DT_KP / 32, dt_proj_b, 1, stream);

    // ---- 5. four directional scans accumulated into y_sum -------------------
    scan_kernel<<<48, 256, 0, stream>>>(xconv, xdbl, dt_full, A_log, y_sum);

    // ---- 6. layernorm + silu(z) gate -> bf16 --------------------------------
    ln_gate_kernel<<<L_TOK, 256, 0, stream>>>(y_sum, xz, ln_g, ln_b, g_bf);

    // ---- 7. out_proj: out = g @ out_proj_w  (WMMA, K=192, N=96) -------------
    launch_gemm(g_bf, DINNER, pw_out, out, DMODEL, nullptr,
                L_TOK, DMODEL, DINNER / 32, nullptr, 0, stream);
}
